// MLPLayer_6442450944690
// MI455X (gfx1250) — matrix operand order repaired and checked by reading, not compiled, not run on hardware
//
#include <hip/hip_runtime.h>
#include <hip/hip_bf16.h>
#include <cstddef>
#include <cstdint>

// ---------------------------------------------------------------------------
// Fused gated-MLP with threshold sparsity for MI455X (gfx1250, wave32, WMMA).
//   Pass 1: up/gate GEMMs (bf16 WMMA, f32->bf16 on the fly) + mask + silu
//           -> act (bf16) in workspace [M x I].  Grid: M-tiles fastest so the
//           concurrent window shares weight tiles; weights loaded NT so the
//           134 MB x matrix stays resident in the 192 MB L2.
//   Pass 2: out = act @ w_down^T (bf16 WMMA). Supertile swizzle: 64 M-tiles
//           (115 MB act slice, L2-resident) x full H sweep; w_down streams NT.
//           act tiles staged into LDS via GLOBAL_LOAD_ASYNC_TO_LDS_B128.
// Roofline: ~1.2 GB HBM traffic @ 23.3 TB/s ~= 52 us; WMMA compute << that.
// ---------------------------------------------------------------------------

typedef __attribute__((ext_vector_type(8)))  float  v8f;
typedef __attribute__((ext_vector_type(16))) __bf16 v16bf;
typedef float v4f_n __attribute__((vector_size(16)));  // native vec for NT builtins

#define TILE_M 64
#define TILE_N 64
#define KC     32   // bf16 WMMA native K

// ---- gfx1250 async global->LDS (ASYNCcnt) availability ---------------------
#if defined(__has_builtin)
#if __has_builtin(__builtin_amdgcn_global_load_async_to_lds_b128)
#define HAVE_ASYNC_LDS 1
#endif
#if __has_builtin(__builtin_amdgcn_s_wait_asynccnt)
#define HAVE_WAIT_ASYNC_BUILTIN 1
#endif
#endif

#if defined(HAVE_ASYNC_LDS)
typedef int v4i_vs __attribute__((vector_size(16)));
typedef __attribute__((address_space(1))) v4i_vs gbl_v4i_t;
typedef __attribute__((address_space(3))) v4i_vs lds_v4i_t;
#endif

__device__ __forceinline__ void wait_async_zero() {
#if defined(HAVE_ASYNC_LDS)
#if defined(HAVE_WAIT_ASYNC_BUILTIN)
  __builtin_amdgcn_s_wait_asynccnt(0);
#else
  asm volatile("s_wait_asynccnt 0x0" ::: "memory");
#endif
#endif
}

// ---- LDS stage: 64x32 tile, f32 source converted to bf16 -------------------
// NT=true  -> TH=NT load policy (streaming operand, don't evict L2 residents)
template <bool NT>
__device__ __forceinline__ void load_tile_f32_bf16(__bf16* __restrict__ dst,
                                                   const float* __restrict__ src,
                                                   int ld, int rowBase, int kBase,
                                                   int tid) {
  const int row = tid >> 2;          // 0..63
  const int col = (tid & 3) << 3;    // 0,8,16,24
  const v4f_n* p =
      reinterpret_cast<const v4f_n*>(src + (size_t)(rowBase + row) * ld + kBase + col);
  v4f_n v0, v1;
  if (NT) {
    v0 = __builtin_nontemporal_load(p);
    v1 = __builtin_nontemporal_load(p + 1);
  } else {
    v0 = p[0];
    v1 = p[1];
  }
  __bf16* d = dst + row * KC + col;
  d[0] = (__bf16)v0[0]; d[1] = (__bf16)v0[1]; d[2] = (__bf16)v0[2]; d[3] = (__bf16)v0[3];
  d[4] = (__bf16)v1[0]; d[5] = (__bf16)v1[1]; d[6] = (__bf16)v1[2]; d[7] = (__bf16)v1[3];
}

// ---- LDS stage: 64x32 bf16 tile; async DMA into LDS when available ---------
__device__ __forceinline__ void load_tile_bf16(__bf16* __restrict__ dst,
                                               const __bf16* __restrict__ src,
                                               int ld, int rowBase, int kBase,
                                               int tid) {
  const int row = tid >> 2;
  const int col = (tid & 3) << 3;
  const __bf16* g = src + (size_t)(rowBase + row) * ld + kBase + col;  // 16B
  __bf16* l = dst + row * KC + col;
#if defined(HAVE_ASYNC_LDS)
  __builtin_amdgcn_global_load_async_to_lds_b128((gbl_v4i_t*)g, (lds_v4i_t*)l,
                                                 /*offset=*/0, /*cpol=*/0);
#else
  *reinterpret_cast<uint4*>(l) = *reinterpret_cast<const uint4*>(g);
#endif
}

// ---- A fragment (16x32 bf16), ISA 7.12.2 layout ----------------------------
// lanes 0-15:  elems 0-7 = K0..7,  elems 8-15 = K16..23   (row M = lane)
// lanes 16-31: elems 0-7 = K8..15, elems 8-15 = K24..31   (row M = lane-16)
__device__ __forceinline__ v16bf load_a(const __bf16* __restrict__ s,
                                        int rowBase, int lane) {
  const int row = rowBase + (lane & 15);
  const int kh  = (lane >> 4) * 8;
  const __bf16* r = s + row * KC;
  v16bf a;
#pragma unroll
  for (int i = 0; i < 8; ++i) a[i] = r[kh + i];
#pragma unroll
  for (int i = 0; i < 8; ++i) a[8 + i] = r[16 + kh + i];
  return a;
}

// ---- B fragment (32x16 bf16): lanes 0-15 -> K0..7,K16..23 of column N=lane,
//      lanes 16-31 -> K8..15,K24..31 of column N=lane-16. LDS tile is [N][K].
__device__ __forceinline__ v16bf load_b(const __bf16* __restrict__ s,
                                        int colBase, int lane) {
  const int col = colBase + (lane & 15);
  const int k0  = (lane >> 4) * 8;
  const __bf16* r = s + col * KC + k0;
  v16bf b;
#pragma unroll
  for (int i = 0; i < 8; ++i) { b[i] = r[i]; b[8 + i] = r[16 + i]; }
  return b;
}

// ===========================================================================
// Pass 1: act[m,i] = silu(gate)*up where mask=|up*avg_gate[i]|>=0.5, else 0
// Grid: (M/64, I/64) -- M-tiles on x (fastest) => concurrent blocks share
// the same weight tiles; x stays L2-resident, weights stream NT.
// ===========================================================================
__global__ __launch_bounds__(256) void ffn_up_gate_kernel(
    const float* __restrict__ x, const float* __restrict__ w_gate,
    const float* __restrict__ w_up, const float* __restrict__ avg_gate,
    __bf16* __restrict__ act, int M, int H, int I) {
  __shared__ __bf16 sX [2][TILE_M * KC];
  __shared__ __bf16 sWg[2][TILE_N * KC];
  __shared__ __bf16 sWu[2][TILE_N * KC];

  const int tid   = threadIdx.x;
  const int wave  = tid >> 5;
  const int lane  = tid & 31;
  const int mBase = blockIdx.x * TILE_M;   // M fastest
  const int iBase = blockIdx.y * TILE_N;
  const int rowBase = (wave & 3) * 16;     // wave's M sub-tile
  const int colBase = (wave >> 2) * 32;    // wave's N sub-tile (2 x 16)

  const int ldRow = tid >> 2;              // staging row this thread fills
  const int ldCol = (tid & 3) << 3;

  v8f accU[2] = {};
  v8f accG[2] = {};

  const int steps = H >> 5;  // H / KC
  load_tile_f32_bf16<false>(sX[0],  x,      H, mBase, 0, tid);
  load_tile_f32_bf16<true >(sWg[0], w_gate, H, iBase, 0, tid);
  load_tile_f32_bf16<true >(sWu[0], w_up,   H, iBase, 0, tid);
  __syncthreads();

  for (int kk = 0; kk < steps; ++kk) {
    const int cur = kk & 1;
    if (kk + 1 < steps) {
      const int nxt = cur ^ 1;
      const int kB  = (kk + 1) << 5;
      load_tile_f32_bf16<false>(sX[nxt],  x,      H, mBase, kB, tid);
      load_tile_f32_bf16<true >(sWg[nxt], w_gate, H, iBase, kB, tid);
      load_tile_f32_bf16<true >(sWu[nxt], w_up,   H, iBase, kB, tid);
    }
    if (kk + 2 < steps) {  // global_prefetch_b8: warm L2 two stages ahead
      const int kP = (kk + 2) << 5;
      __builtin_prefetch(w_gate + (size_t)(iBase + ldRow) * H + kP + ldCol, 0, 1);
      __builtin_prefetch(w_up   + (size_t)(iBase + ldRow) * H + kP + ldCol, 0, 1);
    }
    v16bf a = load_a(sX[cur], rowBase, lane);
#pragma unroll
    for (int j = 0; j < 2; ++j) {
      v16bf bu = load_b(sWu[cur], colBase + j * 16, lane);
      v16bf bg = load_b(sWg[cur], colBase + j * 16, lane);
      accU[j] = __builtin_amdgcn_wmma_f32_16x16x32_bf16(
          false, a, false, bu, (short)0, accU[j], false, false);
      accG[j] = __builtin_amdgcn_wmma_f32_16x16x32_bf16(
          false, a, false, bg, (short)0, accG[j], false, false);
    }
    __syncthreads();
  }

  // Epilogue: mask + silu, NT-store bf16 activation (streaming output).
  const int mOff = (lane >> 4) * 8;  // C/D layout: upper lanes hold M+8
#pragma unroll
  for (int j = 0; j < 2; ++j) {
    const int n  = colBase + j * 16 + (lane & 15);
    const int ig = iBase + n;
    const float avg = avg_gate[ig];
#pragma unroll
    for (int r = 0; r < 8; ++r) {
      const int m = mBase + rowBase + mOff + r;
      const float up = accU[j][r];
      const float g  = accG[j][r];
      float val = 0.0f;
      if (fabsf(up * avg) >= 0.5f) {
        const float gate = g / (1.0f + __expf(-g));  // silu
        val = gate * up;
      }
      __builtin_nontemporal_store((__bf16)val, &act[(size_t)m * I + ig]);
    }
  }
}

// ===========================================================================
// Pass 2: out[m,h] = sum_i act[m,i] * w_down[h,i]
// Linear grid with supertile swizzle: 64 M-tiles (115 MB act slice resident
// in L2) x full H sweep, M-in-supertile fastest; w_down streams NT.
// ===========================================================================
__global__ __launch_bounds__(256) void ffn_down_kernel(
    const __bf16* __restrict__ act, const float* __restrict__ w_down,
    float* __restrict__ out, int M, int H, int I) {
  __shared__ __bf16 sA[2][TILE_M * KC];
  __shared__ __bf16 sW[2][TILE_N * KC];

  const int tid  = threadIdx.x;
  const int wave = tid >> 5;
  const int lane = tid & 31;

  // Supertile decode
  const int mTiles = M >> 6;
  const int hTiles = H >> 6;
  const int superM = (mTiles < 64) ? mTiles : 64;
  const int perSuper = hTiles * superM;
  const int lin = blockIdx.x;
  const int sm  = lin / perSuper;
  const int rem = lin - sm * perSuper;
  const int mt  = sm * superM + (rem % superM);  // M fastest within supertile
  const int ht  = rem / superM;
  const int mBase = mt * TILE_M;
  const int hBase = ht * TILE_N;

  const int rowBase = (wave & 3) * 16;
  const int colBase = (wave >> 2) * 32;
  const int ldRow = tid >> 2;
  const int ldCol = (tid & 3) << 3;

  v8f acc[2] = {};

  const int steps = I >> 5;  // 448
  load_tile_bf16          (sA[0], act,    I, mBase, 0, tid);  // async ASYNCcnt
  load_tile_f32_bf16<true>(sW[0], w_down, I, hBase, 0, tid);  // NT stream
  wait_async_zero();
  __syncthreads();

  for (int kk = 0; kk < steps; ++kk) {
    const int cur = kk & 1;
    if (kk + 1 < steps) {
      const int nxt = cur ^ 1;
      const int kB  = (kk + 1) << 5;
      load_tile_bf16          (sA[nxt], act,    I, mBase, kB, tid);
      load_tile_f32_bf16<true>(sW[nxt], w_down, I, hBase, kB, tid);
    }
    if (kk + 2 < steps) {
      const int kP = (kk + 2) << 5;
      __builtin_prefetch(w_down + (size_t)(hBase + ldRow) * I + kP + ldCol, 0, 1);
    }
    v16bf a = load_a(sA[cur], rowBase, lane);
#pragma unroll
    for (int j = 0; j < 2; ++j) {
      v16bf b = load_b(sW[cur], colBase + j * 16, lane);
      acc[j] = __builtin_amdgcn_wmma_f32_16x16x32_bf16(
          false, a, false, b, (short)0, acc[j], false, false);
    }
    wait_async_zero();   // next-stage async tile landed before publish barrier
    __syncthreads();
  }

  const int mOff = (lane >> 4) * 8;
#pragma unroll
  for (int j = 0; j < 2; ++j) {
    const int n = colBase + j * 16 + (lane & 15);
#pragma unroll
    for (int r = 0; r < 8; ++r) {
      const int m = mBase + rowBase + mOff + r;
      __builtin_nontemporal_store(acc[j][r], &out[(size_t)m * H + hBase + n]);
    }
  }
}

// ===========================================================================
extern "C" void kernel_launch(void* const* d_in, const int* in_sizes, int n_in,
                              void* d_out, int out_size, void* d_ws, size_t ws_size,
                              hipStream_t stream) {
  const float* x        = (const float*)d_in[0];
  const float* w_gate   = (const float*)d_in[1];
  const float* w_up     = (const float*)d_in[2];
  const float* w_down   = (const float*)d_in[3];
  const float* avg_gate = (const float*)d_in[4];
  float* out = (float*)d_out;

  const int I = in_sizes[4];           // 14336
  const int H = in_sizes[1] / I;       // 4096
  const int M = in_sizes[0] / H;       // 8192 (= B*S)

  __bf16* act = (__bf16*)d_ws;         // M*I*2 bytes = 235 MB bf16 workspace

  dim3 blk(256);
  dim3 g1(M / TILE_M, I / TILE_N);     // 128 x 224, M fastest
  ffn_up_gate_kernel<<<g1, blk, 0, stream>>>(x, w_gate, w_up, avg_gate, act, M, H, I);

  dim3 g2((M / TILE_M) * (H / TILE_N));  // 8192 linear, swizzled in-kernel
  ffn_down_kernel<<<g2, blk, 0, stream>>>(act, w_down, out, M, H, I);
}
